// VisionMambaPathArch_42322607735354
// MI455X (gfx1250) — matrix operand order repaired and checked by reading, not compiled, not run on hardware
//
#include <hip/hip_runtime.h>

// ---------------- model constants ----------------
#define B_SZ    16
#define CIN     3
#define IMGSZ   224
#define PATCH   16
#define D_MODEL 192
#define DEPTH   8
#define N_ST    16
#define K_CONV  4
#define DIN     384          // 2*D
#define R_RANK  12
#define L_PATCH 196
#define SEQ     197          // L + 1 (cls)
#define TPOS    98
#define NCLS    1000
#define EPSF    1e-5f
#define XZC     (2*DIN)      // 768
#define DBLC    (R_RANK + 2*N_ST)  // 44
#define MROWS   (B_SZ*SEQ)   // 3152
#define PROWS   (B_SZ*L_PATCH) // 3136

typedef _Float16 half_t;
typedef __attribute__((ext_vector_type(16))) _Float16 v16h;
typedef __attribute__((ext_vector_type(8)))  _Float16 v8h;
typedef __attribute__((ext_vector_type(8)))  float    v8f;
typedef __attribute__((ext_vector_type(4)))  unsigned int u32x4;
typedef __attribute__((ext_vector_type(8)))  int          i32x8;
typedef __attribute__((ext_vector_type(4)))  int          i32x4;

#define CAT16(l8, h8) __builtin_shufflevector(l8, h8, 0,1,2,3,4,5,6,7,8,9,10,11,12,13,14,15)

#if __has_builtin(__builtin_amdgcn_tensor_load_to_lds)
#define HAVE_TDM 1
#else
#define HAVE_TDM 0
#endif

#if HAVE_TDM
// Issue one TDM 2D tile load (tile0 = contiguous 2B elements, tile1 = rows,
// row stride = stride_elems) into LDS at byte offset lds_off, then drain TENSORcnt.
// D# layout per CDNA5 ISA ch.8 (group0 128b, group1 256b; groups 2/3 zero => <=2D).
__device__ inline void tdm_load_2d(unsigned int lds_off, const void* gptr,
                                   unsigned int tile0_elems, unsigned int tile1_rows,
                                   unsigned long long stride_elems)
{
    unsigned long long ga = (unsigned long long)gptr;
    u32x4 g0;
    g0[0] = 1u;                                        // count=1 (valid), user mode
    g0[1] = lds_off;                                   // lds_addr (bytes)
    g0[2] = (unsigned int)(ga & 0xFFFFFFFFu);          // global_addr[31:0]
    g0[3] = (unsigned int)((ga >> 32) & 0x01FFFFFFu)   // global_addr[56:32]
          | (2u << 30);                                // type = 2 ("image")
    const unsigned int td0 = 0x7FFFFFFFu;              // huge tensor dims (no OOB clip)
    const unsigned int td1 = 0x7FFFFFFFu;
    i32x8 g1;
    g1[0] = (int)(1u << 16);                           // data_size=1 (2 bytes), mask=0
    g1[1] = (int)((td0 & 0xFFFFu) << 16);              // atomic_addr=0 | tensor_dim0 lo
    g1[2] = (int)((td0 >> 16) | ((td1 & 0xFFFFu) << 16));
    g1[3] = (int)((td1 >> 16) | ((tile0_elems & 0xFFFFu) << 16)); // tile_dim0
    g1[4] = (int)(tile1_rows & 0xFFFFu);               // tile_dim1 ; tile_dim2=0
    g1[5] = (int)(unsigned int)(stride_elems & 0xFFFFFFFFu);      // dim0_stride lo32
    g1[6] = (int)(unsigned int)((stride_elems >> 32) & 0xFFFFu);  // dim0_stride hi16
    g1[7] = 0;                                         // dim1_stride = 0 (unused)
    i32x4 gz = {0, 0, 0, 0};
#if __clang_major__ >= 23
    i32x8 gz8 = {};
    __builtin_amdgcn_tensor_load_to_lds(g0, g1, gz, gz, gz8, 0);
#else
    __builtin_amdgcn_tensor_load_to_lds(g0, g1, gz, gz, 0);
#endif
    __builtin_amdgcn_s_wait_tensorcnt(0);
}
#endif

// ---------------- WMMA GEMM v3 ----------------
// D[M,Ncols] = A[M,KDIM](f16, row-major, lda==KDIM) * W[Ncols,KDIM]^T (f16, ldb==KDIM)
//              (+ Cin if non-null), f32 out (ldc runtime).
// Block = 128 threads = 4 waves sharing one 16-row A tile staged in LDS via TDM
// (single 1-D DMA: the tile is contiguous since lda==KDIM). Each wave computes a
// 16x64 strip. Fast path (all 4 n-tiles valid) is branch-free: per K-step
// 2 ds_load_b128 (A) + 8 global_load_b128 (B) + 4 v_wmma.
template <int KDIM, bool MASKED>
__global__ void __launch_bounds__(128)
wmma_gemm_v3(const half_t* __restrict__ A,
             const half_t* __restrict__ Bw,
             const float* __restrict__ Cin, float* __restrict__ Dout,
             int ldc, int Ncols)
{
    __shared__ half_t As[16 * KDIM];
    const int lane   = threadIdx.x & 31;
    const int wave_u = __builtin_amdgcn_readfirstlane(threadIdx.x >> 5); // scalar
    const int m0 = blockIdx.x * 16;
    const int hi = lane >> 4;
    const int lo = lane & 15;

    // ---- stage contiguous A tile (16*KDIM halves) into LDS ----
#if HAVE_TDM
    if (wave_u == 0) {
        tdm_load_2d((unsigned int)(unsigned long long)(As),
                    A + (size_t)m0 * KDIM, (unsigned int)(16 * KDIM), 1u, 0ull);
    }
#else
    for (int idx = threadIdx.x * 8; idx < 16 * KDIM; idx += 128 * 8)
        *(v8h*)(As + idx) = *(const v8h*)(A + (size_t)m0 * KDIM + idx);
#endif
    __syncthreads();

    const int ntiles = (Ncols + 15) >> 4;
    const int ntbase = blockIdx.y * 16 + wave_u * 4;   // scalar
    int ntValid = ntiles - ntbase;                     // scalar
    if (ntValid <= 0) return;
    if (ntValid > 4) ntValid = 4;

    const half_t* aL = As + (size_t)lo * KDIM;
    const v16h zb = {};

    if (ntValid == 4) {
        // ---------- branch-free fast path: 4 tiles ----------
        const half_t* brow[4];
        bool cOK[4];
        #pragma unroll
        for (int t = 0; t < 4; ++t) {
            int col = (ntbase + t) * 16 + lo;
            cOK[t] = (col < Ncols);
            if (MASKED && col >= Ncols) col = Ncols - 1;
            brow[t] = Bw + (size_t)col * KDIM;
        }
        v8f acc[4] = {{}, {}, {}, {}};
        for (int k0 = 0; k0 < KDIM; k0 += 32) {
            // A fragment from LDS: e<8 -> k=8*hi+e ; e>=8 -> k=16+8*hi+(e-8)
            v8h a0 = *(const v8h*)(aL + k0 + 8 * hi);
            v8h a1 = *(const v8h*)(aL + k0 + 16 + 8 * hi);
            v16h a = CAT16(a0, a1);
            #pragma unroll
            for (int t = 0; t < 4; ++t) {
                // B fragment: e<8 -> k=8*hi+e ; e>=8 -> k=16+8*hi+(e-8)
                v8h b0 = *(const v8h*)(brow[t] + k0 + 8 * hi);
                v8h b1 = *(const v8h*)(brow[t] + k0 + 16 + 8 * hi);
                v16h b = CAT16(b0, b1);
                if (MASKED && !cOK[t]) b = zb;
                acc[t] = __builtin_amdgcn_wmma_f32_16x16x32_f16(
                             false, a, false, b, (short)0, acc[t], false, false);
            }
        }
        #pragma unroll
        for (int t = 0; t < 4; ++t) {
            int c = (ntbase + t) * 16 + lo;
            #pragma unroll
            for (int v = 0; v < 8; ++v) {
                if (!MASKED || cOK[t]) {
                    int r = m0 + v + 8 * hi;
                    float val = acc[t][v];
                    if (Cin) val += Cin[(size_t)r * ldc + c];
                    Dout[(size_t)r * ldc + c] = val;
                }
            }
        }
    } else {
        // ---------- remainder path: 1..3 tiles ----------
        for (int t = 0; t < ntValid; ++t) {            // scalar bound
            int col = (ntbase + t) * 16 + lo;
            bool ok = (col < Ncols);
            int colc = (MASKED && !ok) ? (Ncols - 1) : col;
            const half_t* brow = Bw + (size_t)colc * KDIM;
            v8f acc = {};
            for (int k0 = 0; k0 < KDIM; k0 += 32) {
                v8h a0 = *(const v8h*)(aL + k0 + 8 * hi);
                v8h a1 = *(const v8h*)(aL + k0 + 16 + 8 * hi);
                v16h a = CAT16(a0, a1);
                v8h b0 = *(const v8h*)(brow + k0 + 8 * hi);
                v8h b1 = *(const v8h*)(brow + k0 + 16 + 8 * hi);
                v16h b = CAT16(b0, b1);
                if (MASKED && !ok) b = zb;
                acc = __builtin_amdgcn_wmma_f32_16x16x32_f16(
                          false, a, false, b, (short)0, acc, false, false);
            }
            #pragma unroll
            for (int v = 0; v < 8; ++v) {
                if (!MASKED || ok) {
                    int r = m0 + v + 8 * hi;
                    float val = acc[v];
                    if (Cin) val += Cin[(size_t)r * ldc + col];
                    Dout[(size_t)r * ldc + col] = val;
                }
            }
        }
    }
}

// ---------------- helpers ----------------
__device__ inline float blockReduceSum256(float v) {
    __shared__ float s[256];
    int t = threadIdx.x;
    s[t] = v;
    __syncthreads();
    for (int off = 128; off > 0; off >>= 1) {
        if (t < off) s[t] += s[t + off];
        __syncthreads();
    }
    return s[0];
}

__device__ inline float siluf(float x) { return x / (1.f + expf(-x)); }

// ---------------- elementwise / prep kernels ----------------
__global__ void convert_f32_f16_kernel(const float* __restrict__ src,
                                       half_t* __restrict__ dst, int n) {
    int t = blockIdx.x * blockDim.x + threadIdx.x;
    if (t < n) dst[t] = (half_t)src[t];
}

// im2col for the 16x16/stride16 patch conv: A[b*196+p][c*256+ph*16+pw]
__global__ void im2col_kernel(const float* __restrict__ x, half_t* __restrict__ ic) {
    int t = blockIdx.x * blockDim.x + threadIdx.x;
    if (t >= PROWS * (CIN * PATCH * PATCH)) return;
    int k = t % 768;  int rp = t / 768;
    int b = rp / L_PATCH, p = rp % L_PATCH;
    int c = k >> 8;   int r2 = k & 255;
    int ph = r2 >> 4; int pw = r2 & 15;
    int py = p / 14,  px = p % 14;
    float v = x[(((size_t)b * CIN + c) * IMGSZ + (py * PATCH + ph)) * IMGSZ + (px * PATCH + pw)];
    ic[t] = (half_t)v;
}

// patch-GEMM output -> +bias -> rmsnorm(embed_w) -> token insert -> +pos_embed
__global__ void embed_kernel(const float* __restrict__ y0, const float* __restrict__ pb,
                             const float* __restrict__ ew, const float* __restrict__ pos,
                             float* __restrict__ h) {
    int row = blockIdx.x;              // 0..3135
    int d = threadIdx.x;
    int b = row / L_PATCH, p = row % L_PATCH;
    float v = 0.f;
    if (d < D_MODEL) v = y0[(size_t)row * D_MODEL + d] + pb[d];
    float ss = blockReduceSum256(v * v);
    float rinv = rsqrtf(ss / (float)D_MODEL + EPSF);
    int lpos = (p < TPOS) ? p : p + 1;
    if (d < D_MODEL)
        h[((size_t)b * SEQ + lpos) * D_MODEL + d] = ew[d] * v * rinv
                                                  + pos[(size_t)lpos * D_MODEL + d];
}

__global__ void cls_kernel(const float* __restrict__ cls, const float* __restrict__ pos,
                           float* __restrict__ h) {
    int d = threadIdx.x, b = blockIdx.x;
    if (d < D_MODEL)
        h[((size_t)b * SEQ + TPOS) * D_MODEL + d] = cls[d] + pos[(size_t)TPOS * D_MODEL + d];
}

template <bool OUT_HALF>
__global__ void rmsnorm_kernel(const float* __restrict__ x, const float* __restrict__ w,
                               void* __restrict__ out) {
    int row = blockIdx.x;
    int d = threadIdx.x;
    float v = 0.f;
    if (d < D_MODEL) v = x[(size_t)row * D_MODEL + d];
    float ss = blockReduceSum256(v * v);
    float rinv = rsqrtf(ss / (float)D_MODEL + EPSF);
    if (d < D_MODEL) {
        float o = w[d] * v * rinv;
        if (OUT_HALF) ((half_t*)out)[(size_t)row * D_MODEL + d] = (half_t)o;
        else          ((float*)out)[(size_t)row * D_MODEL + d] = o;
    }
}

// causal depthwise conv (K=4) on the u half of xz, +bias, silu -> u (f32 + f16)
__global__ void conv_silu_kernel(const float* __restrict__ xz, const float* __restrict__ cw,
                                 const float* __restrict__ cb, float* __restrict__ u,
                                 half_t* __restrict__ uh) {
    int t = blockIdx.x * blockDim.x + threadIdx.x;
    if (t >= MROWS * DIN) return;
    int c = t % DIN;  int row = t / DIN;
    int b = row / SEQ, l = row % SEQ;
    float s = cb[c];
    #pragma unroll
    for (int k = 0; k < K_CONV; ++k) {
        int ls = l - (K_CONV - 1) + k;
        if (ls >= 0) s += cw[c * K_CONV + k] * xz[((size_t)b * SEQ + ls) * XZC + c];
    }
    float sv = siluf(s);
    u[t] = sv;
    uh[t] = (half_t)sv;
}

// dt = softplus(dbl[:, :12] @ Wdt^T + dtb)
__global__ void dt_kernel(const float* __restrict__ dbl, const float* __restrict__ Wdt,
                          const float* __restrict__ dtb, float* __restrict__ dt) {
    int t = blockIdx.x * blockDim.x + threadIdx.x;
    if (t >= MROWS * DIN) return;
    int d = t % DIN, row = t / DIN;
    float s = dtb[d];
    #pragma unroll
    for (int r = 0; r < R_RANK; ++r)
        s += dbl[(size_t)row * DBLC + r] * Wdt[d * R_RANK + r];
    dt[t] = (s > 20.f) ? s : log1pf(expf(s));
}

// selective scan: one thread per (b,d), N=16 state in registers; fused epilogue
__global__ void scan_kernel(const float* __restrict__ dt, const float* __restrict__ u,
                            const float* __restrict__ dbl, const float* __restrict__ xz,
                            const float* __restrict__ Alog, const float* __restrict__ Dp,
                            half_t* __restrict__ yh) {
    int t = blockIdx.x * blockDim.x + threadIdx.x;
    if (t >= B_SZ * DIN) return;
    int d = t % DIN, b = t / DIN;
    float a[N_ST], st[N_ST];
    #pragma unroll
    for (int n = 0; n < N_ST; ++n) { a[n] = -expf(Alog[d * N_ST + n]); st[n] = 0.f; }
    float dpv = Dp[d];
    for (int l = 0; l < SEQ; ++l) {
        size_t row = (size_t)b * SEQ + l;
        float dtv = dt[row * DIN + d];
        float uv  = u[row * DIN + d];
        const float* brow = dbl + row * DBLC;
        float acc = 0.f;
        #pragma unroll
        for (int n = 0; n < N_ST; ++n) {
            float dA = expf(dtv * a[n]);
            st[n] = dA * st[n] + dtv * brow[R_RANK + n] * uv;
            acc += st[n] * brow[R_RANK + N_ST + n];
        }
        float y = acc + uv * dpv;
        float z = xz[row * XZC + DIN + d];
        y *= siluf(z);
        yh[row * DIN + d] = (half_t)y;
    }
}

// mean over tokens excluding cls (l==TPOS)
__global__ void mean_kernel(const float* __restrict__ hn, half_t* __restrict__ feat) {
    int t = blockIdx.x * blockDim.x + threadIdx.x;
    if (t >= B_SZ * D_MODEL) return;
    int d = t % D_MODEL, b = t / D_MODEL;
    float s = 0.f;
    for (int l = 0; l < SEQ; ++l) {
        if (l == TPOS) continue;
        s += hn[((size_t)b * SEQ + l) * D_MODEL + d];
    }
    feat[b * D_MODEL + d] = (half_t)(s / (float)L_PATCH);
}

__global__ void bias_init_kernel(const float* __restrict__ hb, float* __restrict__ out) {
    int t = blockIdx.x * blockDim.x + threadIdx.x;
    if (t < B_SZ * NCLS) out[t] = hb[t % NCLS];
}

// ---------------- launcher ----------------
extern "C" void kernel_launch(void* const* d_in, const int* in_sizes, int n_in,
                              void* d_out, int out_size, void* d_ws, size_t ws_size,
                              hipStream_t stream) {
    (void)in_sizes; (void)n_in; (void)out_size; (void)ws_size;
    const float* x        = (const float*)d_in[0];
    const float* patch_w  = (const float*)d_in[1];
    const float* patch_b  = (const float*)d_in[2];
    const float* embed_nw = (const float*)d_in[3];
    const float* cls_tok  = (const float*)d_in[4];
    const float* pos_emb  = (const float*)d_in[5];
    const float* norm_w   = (const float*)d_in[6];
    const float* in_proj  = (const float*)d_in[7];
    const float* conv_w   = (const float*)d_in[8];
    const float* conv_b   = (const float*)d_in[9];
    const float* x_proj   = (const float*)d_in[10];
    const float* dt_proj  = (const float*)d_in[11];
    const float* dt_bias  = (const float*)d_in[12];
    const float* A_log    = (const float*)d_in[13];
    const float* D_param  = (const float*)d_in[14];
    const float* out_proj = (const float*)d_in[15];
    const float* final_nw = (const float*)d_in[16];
    const float* head_w   = (const float*)d_in[17];
    const float* head_b   = (const float*)d_in[18];
    float* out = (float*)d_out;

    size_t off = 0;
    auto alloc = [&](size_t bytes) -> void* {
        void* p = (char*)d_ws + off;
        off += (bytes + 255) & ~(size_t)255;
        return p;
    };
    half_t* wPatch = (half_t*)alloc((size_t)D_MODEL * 768 * 2);
    half_t* wIn    = (half_t*)alloc((size_t)DEPTH * XZC * D_MODEL * 2);
    half_t* wX     = (half_t*)alloc((size_t)DEPTH * DBLC * DIN * 2);
    half_t* wOut   = (half_t*)alloc((size_t)DEPTH * D_MODEL * DIN * 2);
    half_t* wHead  = (half_t*)alloc((size_t)NCLS * D_MODEL * 2);
    half_t* imcol  = (half_t*)alloc((size_t)PROWS * 768 * 2);
    float*  y0     = (float*) alloc((size_t)PROWS * D_MODEL * 4);
    float*  h      = (float*) alloc((size_t)MROWS * D_MODEL * 4);
    half_t* xnh    = (half_t*)alloc((size_t)MROWS * D_MODEL * 2);
    float*  xz     = (float*) alloc((size_t)MROWS * XZC * 4);
    float*  u      = (float*) alloc((size_t)MROWS * DIN * 4);
    half_t* uh     = (half_t*)alloc((size_t)MROWS * DIN * 2);
    float*  dbl    = (float*) alloc((size_t)MROWS * DBLC * 4);
    float*  dtbuf  = (float*) alloc((size_t)MROWS * DIN * 4);
    half_t* yhh    = (half_t*)alloc((size_t)MROWS * DIN * 2);
    float*  tmpn   = (float*) alloc((size_t)MROWS * D_MODEL * 4);
    half_t* feat   = (half_t*)alloc((size_t)B_SZ * D_MODEL * 2);

    auto cvt = [&](const float* s, half_t* d, int n) {
        convert_f32_f16_kernel<<<(n + 255) / 256, 256, 0, stream>>>(s, d, n);
    };
    auto gemm = [&](const half_t* A, const half_t* Bw, const float* Cin, float* Dout,
                    int ldc, int M, int Nc, int Kd) {
        int ntiles = (Nc + 15) / 16;
        dim3 grid(M / 16, (ntiles + 15) / 16);
        bool masked = (Nc & 15) != 0;
        if (Kd == 192) {
            if (masked) wmma_gemm_v3<192, true ><<<grid, 128, 0, stream>>>(A, Bw, Cin, Dout, ldc, Nc);
            else        wmma_gemm_v3<192, false><<<grid, 128, 0, stream>>>(A, Bw, Cin, Dout, ldc, Nc);
        } else if (Kd == 384) {
            if (masked) wmma_gemm_v3<384, true ><<<grid, 128, 0, stream>>>(A, Bw, Cin, Dout, ldc, Nc);
            else        wmma_gemm_v3<384, false><<<grid, 128, 0, stream>>>(A, Bw, Cin, Dout, ldc, Nc);
        } else {
            if (masked) wmma_gemm_v3<768, true ><<<grid, 128, 0, stream>>>(A, Bw, Cin, Dout, ldc, Nc);
            else        wmma_gemm_v3<768, false><<<grid, 128, 0, stream>>>(A, Bw, Cin, Dout, ldc, Nc);
        }
    };

    // weight conversion (f32 -> f16), all K-contiguous row-major already
    cvt(patch_w,  wPatch, D_MODEL * 768);
    cvt(in_proj,  wIn,    DEPTH * XZC * D_MODEL);
    cvt(x_proj,   wX,     DEPTH * DBLC * DIN);
    cvt(out_proj, wOut,   DEPTH * D_MODEL * DIN);
    cvt(head_w,   wHead,  NCLS * D_MODEL);

    // patch embed: im2col + WMMA GEMM
    {
        int n = PROWS * 768;
        im2col_kernel<<<(n + 255) / 256, 256, 0, stream>>>(x, imcol);
    }
    gemm(imcol, wPatch, nullptr, y0, D_MODEL, PROWS, D_MODEL, 768);
    embed_kernel<<<PROWS, 256, 0, stream>>>(y0, patch_b, embed_nw, pos_emb, h);
    cls_kernel<<<B_SZ, 192, 0, stream>>>(cls_tok, pos_emb, h);

    // mamba blocks
    for (int i = 0; i < DEPTH; ++i) {
        rmsnorm_kernel<true><<<MROWS, 256, 0, stream>>>(h, norm_w + i * D_MODEL, xnh);
        gemm(xnh, wIn + (size_t)i * XZC * D_MODEL, nullptr, xz, XZC, MROWS, XZC, D_MODEL);
        {
            int n = MROWS * DIN;
            conv_silu_kernel<<<(n + 255) / 256, 256, 0, stream>>>(
                xz, conv_w + i * DIN * K_CONV, conv_b + i * DIN, u, uh);
        }
        gemm(uh, wX + (size_t)i * DBLC * DIN, nullptr, dbl, DBLC, MROWS, DBLC, DIN);
        {
            int n = MROWS * DIN;
            dt_kernel<<<(n + 255) / 256, 256, 0, stream>>>(
                dbl, dt_proj + i * DIN * R_RANK, dt_bias + i * DIN, dtbuf);
        }
        {
            int n = B_SZ * DIN;
            scan_kernel<<<(n + 255) / 256, 256, 0, stream>>>(
                dtbuf, u, dbl, xz, A_log + (size_t)i * DIN * N_ST, D_param + i * DIN, yhh);
        }
        // out_proj with residual accumulate into h (Cin == Dout == h)
        gemm(yhh, wOut + (size_t)i * D_MODEL * DIN, h, h, D_MODEL, MROWS, D_MODEL, DIN);
    }

    // final norm -> masked mean -> head (bias preloaded, WMMA accumulate)
    rmsnorm_kernel<false><<<MROWS, 256, 0, stream>>>(h, final_nw, tmpn);
    {
        int n = B_SZ * D_MODEL;
        mean_kernel<<<(n + 255) / 256, 256, 0, stream>>>(tmpn, feat);
    }
    {
        int n = B_SZ * NCLS;
        bias_init_kernel<<<(n + 255) / 256, 256, 0, stream>>>(head_b, out);
    }
    gemm(feat, wHead, out, out, NCLS, B_SZ, NCLS, D_MODEL);
}
